// NeuronMemory_21157008900536
// MI455X (gfx1250) — matrix operand order repaired and checked by reading, not compiled, not run on hardware
//
#include <hip/hip_runtime.h>
#include <hip/hip_bf16.h>

#define D_MODEL 1024
#define RANK 128
#define N_KNOWLEDGE 32768
#define TOPK_COMPRESS 16
#define KK 8
#define BB 2
#define SS 2048

typedef __bf16 bf16_t;
typedef __bf16 v4bf  __attribute__((ext_vector_type(4)));
typedef __bf16 v8bf  __attribute__((ext_vector_type(8)));
typedef __bf16 v16bf __attribute__((ext_vector_type(16)));
typedef float  v2f   __attribute__((ext_vector_type(2)));
typedef float  v4f   __attribute__((ext_vector_type(4)));
typedef float  v8f   __attribute__((ext_vector_type(8)));

// d_out layout: [B*S*D] output | [B*S*8] topk_idx (as float) | [B*S*8] weights
static constexpr size_t OUT_D   = (size_t)BB * SS * D_MODEL;      // 4194304
static constexpr size_t IDX_OFF = OUT_D;
static constexpr size_t W_OFF   = OUT_D + (size_t)BB * SS * KK;

// ---------------------------------------------------------------------------
// 1) shared_compress[b,d,r] = sum_k w[b,k] * compress_neurons[idx[b,k],d,r]
// ---------------------------------------------------------------------------
__global__ __launch_bounds__(256)
void nm_combine_neurons(const float* __restrict__ w, const int* __restrict__ idx,
                        const float* __restrict__ cn, float* __restrict__ sharedC) {
    int t  = blockIdx.x * 256 + threadIdx.x;           // 0 .. B*D*R-1
    int b  = t / (D_MODEL * RANK);
    int dr = t - b * (D_MODEL * RANK);
    float acc = 0.f;
#pragma unroll
    for (int k = 0; k < TOPK_COMPRESS; ++k) {
        float wk = w[b * TOPK_COMPRESS + k];
        int   nk = idx[b * TOPK_COMPRESS + k];
        acc += wk * cn[(size_t)nk * (D_MODEL * RANK) + dr];
    }
    sharedC[t] = acc;
}

// ---------------------------------------------------------------------------
// 2) knowledge_K fp32 -> bf16 (one-time cast; keeps big GEMM on bf16 WMMA)
// ---------------------------------------------------------------------------
__global__ __launch_bounds__(256)
void nm_cast_K(const float* __restrict__ Kf, bf16_t* __restrict__ Kb) {
    int t = blockIdx.x * 256 + threadIdx.x;            // each handles 4 elems
    v4f f = *(const v4f*)(Kf + (size_t)t * 4);
    v4bf o;
    o[0] = (bf16_t)f[0]; o[1] = (bf16_t)f[1];
    o[2] = (bf16_t)f[2]; o[3] = (bf16_t)f[3];
    *(v4bf*)(Kb + (size_t)t * 4) = o;
}

// ---------------------------------------------------------------------------
// 3) Q[b,s,r] = sum_d x[b,s,d] * shared[b,d,r]  — fp32 WMMA 16x16x4,
//    result stored as bf16 for the scores GEMM. One wave per 16x16 tile.
// ---------------------------------------------------------------------------
__global__ __launch_bounds__(256)
void nm_q_gemm(const float* __restrict__ x, const float* __restrict__ sharedC,
               bf16_t* __restrict__ Qb) {
    const int wid  = blockIdx.x * 8 + (threadIdx.x >> 5);
    const int lane = threadIdx.x & 31;
    const int tilesPerB = (SS / 16) * (RANK / 16);     // 1024
    const int b     = wid / tilesPerB;
    const int rem   = wid - b * tilesPerB;
    const int s0    = (rem / (RANK / 16)) * 16;
    const int r0    = (rem % (RANK / 16)) * 16;
    const int l15   = lane & 15;
    const int hi    = lane >> 4;

    // A (16s x 4d): lane<16 holds {K0,K1}, lane>=16 holds {K2,K3} of row M=l15
    const float* xrow = x + ((size_t)b * SS + s0 + l15) * D_MODEL + hi * 2;
    // B (4d x 16r): lane<16 holds {K0,K1}, lane>=16 holds {K2,K3} of col N=l15
    const float* srow = sharedC + (size_t)b * D_MODEL * RANK + (size_t)(hi * 2) * RANK + r0 + l15;

    v8f c = {};
    for (int d0 = 0; d0 < D_MODEL; d0 += 4) {
        v2f a = *(const v2f*)(xrow + d0);
        v2f bv;
        bv[0] = srow[(size_t)d0 * RANK];
        bv[1] = srow[(size_t)(d0 + 1) * RANK];
        c = __builtin_amdgcn_wmma_f32_16x16x4_f32(false, a, false, bv,
                                                  (short)0, c, false, false);
    }
    // C layout: lane l15 (+hi) holds col r=r0+l15, rows s = s0 + hi*8 + v
    bf16_t* qout = Qb + ((size_t)b * SS + s0 + hi * 8) * RANK + r0 + l15;
#pragma unroll
    for (int v = 0; v < 8; ++v)
        qout[(size_t)v * RANK] = (bf16_t)c[v];
}

// ---------------------------------------------------------------------------
// 4) Fused: scores (bf16 WMMA, f32 acc) -> per-lane top-8 -> LDS merge ->
//    softmax -> weighted V gather.
//    One 512-thread block per (b, 32-token tile). Each streamed A tile
//    (knowledge_K rows) feeds TWO register-resident B operand sets
//    (tokens 0..15 and 16..31), so the hot loop is 8 WMMA : 8 b128 loads
//    and K-side L2 traffic is halved vs a 16-token tile.
//    16 waves split N=32768 (2048 n each); scores never touch memory.
// ---------------------------------------------------------------------------
__global__ __launch_bounds__(512)
void nm_scores_topk_out(const bf16_t* __restrict__ Qb, const bf16_t* __restrict__ Kb,
                        const float* __restrict__ V, float* __restrict__ out) {
    __shared__ float          ls[32][257];    // candidate scores (padded)
    __shared__ unsigned short li[32][257];    // candidate indices (N<65536)
    __shared__ float fw[32][KK];
    __shared__ int   fi[32][KK];

    const int tid  = threadIdx.x;
    const int lane = tid & 31;
    const int wave = tid >> 5;                // 0..15
    const int l15  = lane & 15;
    const int hi   = lane >> 4;

    const int b  = blockIdx.x / (SS / 32);
    const int s0 = (blockIdx.x % (SS / 32)) * 32;

    // Two B operand sets (32x16 each): lane = token column, 16 contiguous r.
    const bf16_t* qrowA = Qb + ((size_t)b * SS + s0 + l15) * RANK + hi * 16;
    const bf16_t* qrowB = qrowA + (size_t)16 * RANK;
    v16bf bqA[4], bqB[4];
#pragma unroll
    for (int j = 0; j < 4; ++j) {
        bqA[j] = *(const v16bf*)(qrowA + 32 * j);
        bqB[j] = *(const v16bf*)(qrowB + 32 * j);
    }

    float tsA[8], tsB[8]; int tiA[8], tiB[8];
#pragma unroll
    for (int j = 0; j < 8; ++j) {
        tsA[j] = -3.4e38f; tiA[j] = 0;
        tsB[j] = -3.4e38f; tiB[j] = 0;
    }

    const float SCALE = 0.08838834764831845f;          // 1/sqrt(128)
    const int nStart = wave * (N_KNOWLEDGE / 16);      // 2048 n per wave
    const int nEnd   = nStart + (N_KNOWLEDGE / 16);

    for (int n0 = nStart; n0 < nEnd; n0 += 16) {
        // A (16n x 32r bf16): lane row M = l15, K-halves split by hi.
        const bf16_t* krow = Kb + (size_t)(n0 + l15) * RANK + hi * 16;
        __builtin_prefetch(krow + (size_t)8 * 16 * RANK, 0, 1);   // 8 tiles ahead
        v8f cA = {}, cB = {};
#pragma unroll
        for (int j = 0; j < 4; ++j) {
            v8bf klo = *(const v8bf*)(krow + 32 * j);
            v8bf khi = *(const v8bf*)(krow + 32 * j + 8);
            v16bf a = __builtin_shufflevector(klo, khi,
                0, 1, 2, 3, 4, 5, 6, 7, 8, 9, 10, 11, 12, 13, 14, 15);
            cA = __builtin_amdgcn_wmma_f32_16x16x32_bf16(false, a, false, bqA[j],
                                                         (short)0, cA, false, false);
            cB = __builtin_amdgcn_wmma_f32_16x16x32_bf16(false, a, false, bqB[j],
                                                         (short)0, cB, false, false);
        }
        // C: lane owns tokens s0+l15 (cA) and s0+16+l15 (cB); n = n0+hi*8+v
        const int nb = n0 + hi * 8;
#pragma unroll
        for (int v = 0; v < 8; ++v) {
            float sv = cA[v] * SCALE;
            if (sv > tsA[7]) {                          // rare path
                float cv = sv; int ci = nb + v;
#pragma unroll
                for (int j = 0; j < 8; ++j) {
                    if (cv > tsA[j]) {
                        float tf = tsA[j]; int td = tiA[j];
                        tsA[j] = cv; tiA[j] = ci; cv = tf; ci = td;
                    }
                }
            }
            float sw = cB[v] * SCALE;
            if (sw > tsB[7]) {
                float cv = sw; int ci = nb + v;
#pragma unroll
                for (int j = 0; j < 8; ++j) {
                    if (cv > tsB[j]) {
                        float tf = tsB[j]; int td = tiB[j];
                        tsB[j] = cv; tiB[j] = ci; cv = tf; ci = td;
                    }
                }
            }
        }
    }

    // 32 candidate lists of 8 per token -> LDS
    const int slot = (wave * 2 + hi) * 8;
#pragma unroll
    for (int j = 0; j < 8; ++j) {
        ls[l15][slot + j]      = tsA[j];
        li[l15][slot + j]      = (unsigned short)tiA[j];
        ls[16 + l15][slot + j] = tsB[j];
        li[16 + l15][slot + j] = (unsigned short)tiB[j];
    }
    __syncthreads();

    if (tid < 32) {                                     // one thread per token
        float ms[8]; int mi[8];
#pragma unroll
        for (int j = 0; j < 8; ++j) { ms[j] = -3.4e38f; mi[j] = 0; }
        for (int q = 0; q < 256; ++q) {
            float cv = ls[tid][q]; int ci = (int)li[tid][q];
            if (cv > ms[7]) {
#pragma unroll
                for (int j = 0; j < 8; ++j) {
                    if (cv > ms[j]) {
                        float tf = ms[j]; int td = mi[j];
                        ms[j] = cv; mi[j] = ci; cv = tf; ci = td;
                    }
                }
            }
        }
        float mx = ms[0];
        float es[8]; float ssum = 0.f;
#pragma unroll
        for (int j = 0; j < 8; ++j) { es[j] = __expf(ms[j] - mx); ssum += es[j]; }
        float inv = 1.f / ssum;
        size_t row = (size_t)b * SS + s0 + tid;
#pragma unroll
        for (int j = 0; j < 8; ++j) {
            float wgt = es[j] * inv;
            fw[tid][j] = wgt; fi[tid][j] = mi[j];
            out[IDX_OFF + row * KK + j] = (float)mi[j];
            out[W_OFF   + row * KK + j] = wgt;
        }
    }
    __syncthreads();

    // out[b,s,:] = sum_k w_k * V[idx_k,:]   (float4, 512 threads over 32x1024)
    for (int it = tid; it < 32 * (D_MODEL / 4); it += 512) {
        int sl = it >> 8;                               // D/4 == 256
        int d4 = it & 255;
        v4f acc = {0.f, 0.f, 0.f, 0.f};
#pragma unroll
        for (int k = 0; k < KK; ++k) {
            float wgt = fw[sl][k];
            v4f vv = *(const v4f*)(V + (size_t)fi[sl][k] * D_MODEL + d4 * 4);
            acc += wgt * vv;
        }
        *(v4f*)(out + ((size_t)b * SS + s0 + sl) * D_MODEL + d4 * 4) = acc;
    }
}

// ---------------------------------------------------------------------------
extern "C" void kernel_launch(void* const* d_in, const int* in_sizes, int n_in,
                              void* d_out, int out_size, void* d_ws, size_t ws_size,
                              hipStream_t stream) {
    const float* x    = (const float*)d_in[0];
    const float* mw   = (const float*)d_in[1];
    const int*   midx = (const int*)d_in[2];
    const float* cn   = (const float*)d_in[3];
    const float* kK   = (const float*)d_in[4];
    const float* kV   = (const float*)d_in[5];
    float* out = (float*)d_out;

    char* ws = (char*)d_ws;
    float*  sharedC = (float*)ws;                       // 1 MB  [B,D,R] f32
    bf16_t* Qb      = (bf16_t*)(ws + (1u << 20));       // 1 MB  [B,S,R] bf16
    bf16_t* Kb      = (bf16_t*)(ws + (2u << 20));       // 8 MB  [N,R]   bf16

    nm_combine_neurons<<<(BB * D_MODEL * RANK) / 256, 256, 0, stream>>>(mw, midx, cn, sharedC);
    nm_cast_K<<<(N_KNOWLEDGE * RANK / 4) / 256, 256, 0, stream>>>(kK, Kb);
    nm_q_gemm<<<(BB * (SS / 16) * (RANK / 16)) / 8, 256, 0, stream>>>(x, sharedC, Qb);
    nm_scores_topk_out<<<BB * (SS / 32), 512, 0, stream>>>(Qb, Kb, kV, out);
}
